// FlashAttn_20753281974699
// MI455X (gfx1250) — matrix operand order repaired and checked by reading, not compiled, not run on hardware
//
#include <hip/hip_runtime.h>
#include <hip/hip_bf16.h>
#include <stdint.h>
#include <stddef.h>

// ---------------------------------------------------------------------------
// Types for CDNA5 WMMA (wave32): V_WMMA_F32_16X16X32_BF16
// ---------------------------------------------------------------------------
typedef __attribute__((ext_vector_type(16))) __bf16        v16bf;
typedef __attribute__((ext_vector_type(8)))  float         v8f;
typedef __attribute__((ext_vector_type(4)))  unsigned int  u32x4;
typedef __attribute__((ext_vector_type(2)))  unsigned int  u32x2;
typedef __attribute__((ext_vector_type(4)))  float         f32x4;

#define EMBED 1024
#define HEADS 16
#define HDIM  64
#define SEQ   2048
#define BATCH 2
#define MTOT  (BATCH * SEQ)   // 4096 rows
#define QKVN  (3 * EMBED)     // 3072 cols

// Padded LDS leading dims (elements). Multiple of 8 so every per-lane
// fragment slice is 16-byte aligned -> ds_load_b128.
#define LDK32 40   // tiles with 32-wide minor dim
#define LDK64 72   // tiles with 64-wide minor dim

union BF16x16 { u32x4 u[2]; v16bf v; };
union Pack4   { u32x2 u; __bf16 h[4]; };
union Pack2   { unsigned int u; __bf16 h[2]; };

// ---------------------------------------------------------------------------
// Fragment loaders (layouts per CDNA5 ISA 7.12.2, wave32), two 8-element runs 16 apart
// per-lane -> two ds_load_b128 each.
// ---------------------------------------------------------------------------
static __device__ __forceinline__ v16bf load_a_frag(const __bf16* lds, int ld,
                                                    int row_base, int k_base, int lane) {
  const __bf16* p = lds + (row_base + (lane & 15)) * ld + k_base + ((lane & 16) ? 8 : 0);
  BF16x16 f;
  f.u[0] = *(const u32x4*)(p);        // regs 0-3
  f.u[1] = *(const u32x4*)(p + 16);   // regs 4-7
  return f.v;
}

// B fragment where LDS stores the operand transposed: lds[n][k], need B[k][n].
static __device__ __forceinline__ v16bf load_bT_frag(const __bf16* lds, int ld,
                                                     int k_base, int col_base, int lane) {
  const __bf16* p = lds + (col_base + (lane & 15)) * ld + k_base + ((lane & 16) ? 8 : 0);
  BF16x16 f;
  f.u[0] = *(const u32x4*)(p);
  f.u[1] = *(const u32x4*)(p + 16);
  return f.v;
}

static __device__ __forceinline__ v8f wmma_bf16(v16bf a, v16bf b, v8f c) {
  return __builtin_amdgcn_wmma_f32_16x16x32_bf16(false, a, false, b, (short)0, c, false, false);
}

static __device__ __forceinline__ v8f vzero8() {
  v8f z = {0.f, 0.f, 0.f, 0.f, 0.f, 0.f, 0.f, 0.f};
  return z;
}

// ---------------------------------------------------------------------------
// Kernel 1: QKV GEMM.  x[4096,1024] f32 @ W_qkv[1024,3072] f32 + b_qkv
// -> Q (x 1/sqrt(64)), K, V bf16 in [B,H,N,D] layout.
// 128 threads (4 waves), 64x64 tile, K-steps of 32.
// Routing (q/k/v select, head index) is uniform per block since the 64-wide
// column tile never crosses a 1024 or 64 boundary.
// ---------------------------------------------------------------------------
__global__ __launch_bounds__(128)
void qkv_gemm_kernel(const float* __restrict__ x, const float* __restrict__ Wqkv,
                     const float* __restrict__ bqkv,
                     __bf16* __restrict__ qg, __bf16* __restrict__ kg,
                     __bf16* __restrict__ vg) {
  __shared__ alignas(16) __bf16 As[64 * LDK32];   // [row][k]
  __shared__ alignas(16) __bf16 Bs[64 * LDK32];   // transposed: [n][k]
  const int tid  = threadIdx.x;
  const int lane = tid & 31;
  const int wave = tid >> 5;
  const int m0 = blockIdx.y * 64;
  const int n0 = blockIdx.x * 64;
  const int wr = (wave >> 1) * 32;
  const int wc = (wave & 1) * 32;

  v8f acc[2][2];
#pragma unroll
  for (int i = 0; i < 2; ++i)
#pragma unroll
    for (int j = 0; j < 2; ++j) acc[i][j] = vzero8();

  for (int k0 = 0; k0 < EMBED; k0 += 32) {
    __syncthreads();
    // A tile 64x32: float4 loads, packed 4xbf16 stores
#pragma unroll
    for (int i = 0; i < 4; ++i) {
      int e = tid + i * 128;             // 0..511 quads
      int r = e >> 3, q = (e & 7) * 4;
      f32x4 f = *(const f32x4*)(x + (size_t)(m0 + r) * EMBED + k0 + q);
      Pack4 pk;
      pk.h[0] = (__bf16)f[0]; pk.h[1] = (__bf16)f[1];
      pk.h[2] = (__bf16)f[2]; pk.h[3] = (__bf16)f[3];
      *(u32x2*)(&As[r * LDK32 + q]) = pk.u;
    }
    // B tile 32x64 -> transposed Bs[n][k]; pair along k -> packed b32 stores
#pragma unroll
    for (int i = 0; i < 8; ++i) {
      int e = tid + i * 128;             // 0..1023
      int k = (e >> 6) * 2, n = e & 63;
      const float* wp = Wqkv + (size_t)(k0 + k) * QKVN + n0 + n;
      Pack2 w;
      w.h[0] = (__bf16)wp[0];
      w.h[1] = (__bf16)wp[QKVN];
      *(unsigned int*)(&Bs[n * LDK32 + k]) = w.u;
    }
    __syncthreads();

    v16bf a0 = load_a_frag(As, LDK32, wr + 0, 0, lane);
    v16bf a1 = load_a_frag(As, LDK32, wr + 16, 0, lane);
    v16bf b0 = load_bT_frag(Bs, LDK32, 0, wc + 0, lane);
    v16bf b1 = load_bT_frag(Bs, LDK32, 0, wc + 16, lane);
    acc[0][0] = wmma_bf16(a0, b0, acc[0][0]);
    acc[0][1] = wmma_bf16(a0, b1, acc[0][1]);
    acc[1][0] = wmma_bf16(a1, b0, acc[1][0]);
    acc[1][1] = wmma_bf16(a1, b1, acc[1][1]);
  }

  // Uniform per block: which of q/k/v, which head, and the scale.
  const int part = n0 >> 10;              // 0=q 1=k 2=v
  const int h    = (n0 & 1023) >> 6;
  __bf16* dst = (part == 0) ? qg : (part == 1) ? kg : vg;
  const float scl = (part == 0) ? 0.125f : 1.0f;   // HEAD_DIM^-0.5 on Q
  const int colq = lane & 15;
  const float bias0 = bqkv[n0 + wc + colq];
  const float bias1 = bqkv[n0 + wc + 16 + colq];

#pragma unroll
  for (int i = 0; i < 2; ++i)
#pragma unroll
    for (int j = 0; j < 2; ++j)
#pragma unroll
      for (int r = 0; r < 8; ++r) {
        int row = m0 + wr + 16 * i + r + ((lane & 16) ? 8 : 0);
        int d   = wc + 16 * j + colq;
        int b = row >> 11, n = row & (SEQ - 1);
        float v = (acc[i][j][r] + (j ? bias1 : bias0)) * scl;
        dst[(((size_t)(b * HEADS + h)) * SEQ + n) * HDIM + d] = (__bf16)v;
      }
}

// ---------------------------------------------------------------------------
// Kernel 2: flash attention. Block = 64 query rows of one (b,h); 4 waves,
// each wave owns 16 rows. KV streamed in 32-row chunks. Online softmax f32.
// ---------------------------------------------------------------------------
__global__ __launch_bounds__(128)
void attn_kernel(const __bf16* __restrict__ qg, const __bf16* __restrict__ kg,
                 const __bf16* __restrict__ vg, __bf16* __restrict__ ctx) {
  __shared__ alignas(16) __bf16 Qs[64 * LDK64];     // [n][d]
  __shared__ alignas(16) __bf16 Ks[32 * LDK64];     // [m][d] (bT for S)
  __shared__ alignas(16) __bf16 Vst[64 * LDK32];    // transposed: [d][m]
  __shared__ alignas(16) __bf16 Ps[4][16 * LDK32];  // per-wave P [row][m]

  const int tid  = threadIdx.x;
  const int lane = tid & 31;
  const int wave = tid >> 5;
  const int bh   = blockIdx.y;
  const int qb   = blockIdx.x * 64;
  const size_t base = (size_t)bh * SEQ * HDIM;
  const int b = bh >> 4, h = bh & 15;

  // Q tile 64x64 bf16, 8-byte packed copies
#pragma unroll
  for (int i = 0; i < 8; ++i) {
    int e = tid + i * 128;               // 0..1023 quads
    int n = e >> 4, q = (e & 15) * 4;
    *(u32x2*)(&Qs[n * LDK64 + q]) =
        *(const u32x2*)(qg + base + (size_t)(qb + n) * HDIM + q);
  }
  __syncthreads();
  v16bf qa0 = load_a_frag(Qs, LDK64, wave * 16, 0, lane);
  v16bf qa1 = load_a_frag(Qs, LDK64, wave * 16, 32, lane);

  float mrow[8], lrow[8];
#pragma unroll
  for (int r = 0; r < 8; ++r) { mrow[r] = -1e30f; lrow[r] = 0.0f; }
  v8f o[4];
#pragma unroll
  for (int t = 0; t < 4; ++t) o[t] = vzero8();

  const int rl_off = (lane & 16) ? 8 : 0;
  const int colq = lane & 15;
  __bf16* myP = Ps[wave];

  for (int c = 0; c < SEQ; c += 32) {
    __syncthreads();
#pragma unroll
    for (int i = 0; i < 4; ++i) {        // K chunk 32x64 row-major
      int e = tid + i * 128;             // 0..511 quads
      int m = e >> 4, q = (e & 15) * 4;
      *(u32x2*)(&Ks[m * LDK64 + q]) =
          *(const u32x2*)(kg + base + (size_t)(c + m) * HDIM + q);
    }
    // V chunk 32x64 -> transposed [d][m]; pair along m -> packed b32 stores
#pragma unroll
    for (int i = 0; i < 2; ++i) {
      int e = tid + i * 128;             // 0..255
      int m = (e >> 4) * 2, q = (e & 15) * 4;
      Pack4 r0, r1;
      r0.u = *(const u32x2*)(vg + base + (size_t)(c + m) * HDIM + q);
      r1.u = *(const u32x2*)(vg + base + (size_t)(c + m + 1) * HDIM + q);
#pragma unroll
      for (int j = 0; j < 4; ++j) {
        Pack2 w;
        w.h[0] = r0.h[j]; w.h[1] = r1.h[j];
        *(unsigned int*)(&Vst[(q + j) * LDK32 + m]) = w.u;
      }
    }
    __syncthreads();

    // S = Q * K^T  (two 16x16 col tiles, K-dim 64 = 2 WMMAs each)
    v8f s0 = vzero8(), s1 = vzero8();
    s0 = wmma_bf16(qa0, load_bT_frag(Ks, LDK64, 0,  0, lane), s0);
    s0 = wmma_bf16(qa1, load_bT_frag(Ks, LDK64, 32, 0, lane), s0);
    s1 = wmma_bf16(qa0, load_bT_frag(Ks, LDK64, 0,  16, lane), s1);
    s1 = wmma_bf16(qa1, load_bT_frag(Ks, LDK64, 32, 16, lane), s1);

    // online softmax; rows live on 16-lane half groups
#pragma unroll
    for (int r = 0; r < 8; ++r) {
      float v0 = s0[r], v1 = s1[r];
      float mx = fmaxf(v0, v1);
#pragma unroll
      for (int msk = 1; msk < 16; msk <<= 1) mx = fmaxf(mx, __shfl_xor(mx, msk, 32));
      float newm  = fmaxf(mrow[r], mx);
      float scale = __expf(mrow[r] - newm);
      float p0 = __expf(v0 - newm);
      float p1 = __expf(v1 - newm);
      float sum = p0 + p1;
#pragma unroll
      for (int msk = 1; msk < 16; msk <<= 1) sum += __shfl_xor(sum, msk, 32);
      lrow[r] = lrow[r] * scale + sum;
      mrow[r] = newm;
#pragma unroll
      for (int t = 0; t < 4; ++t) o[t][r] *= scale;
      int rl = r + rl_off;
      myP[rl * LDK32 + colq]      = (__bf16)p0;
      myP[rl * LDK32 + 16 + colq] = (__bf16)p1;
    }

    // O += P * V  (P 16x32 A-frag via per-wave LDS; V^T gives 4 col tiles)
    v16bf pa = load_a_frag(myP, LDK32, 0, 0, lane);
#pragma unroll
    for (int t = 0; t < 4; ++t)
      o[t] = wmma_bf16(pa, load_bT_frag(Vst, LDK32, 0, 16 * t, lane), o[t]);
  }

  // normalize and write ctx[B,N,E] bf16 (E index = h*64 + d)
#pragma unroll
  for (int r = 0; r < 8; ++r) {
    float inv = 1.0f / lrow[r];
    int n = qb + wave * 16 + r + rl_off;
#pragma unroll
    for (int t = 0; t < 4; ++t) {
      int e = h * HDIM + 16 * t + colq;
      ctx[(size_t)(b * SEQ + n) * EMBED + e] = (__bf16)(o[t][r] * inv);
    }
  }
}

// ---------------------------------------------------------------------------
// Kernel 3: output projection. ctx[4096,1024] bf16 @ W_proj[1024,1024] f32
// + b_proj -> out f32.
// ---------------------------------------------------------------------------
__global__ __launch_bounds__(128)
void proj_gemm_kernel(const __bf16* __restrict__ ctx, const float* __restrict__ Wp,
                      const float* __restrict__ bp, float* __restrict__ out) {
  __shared__ alignas(16) __bf16 As[64 * LDK32];   // [row][k]
  __shared__ alignas(16) __bf16 Bs[64 * LDK32];   // transposed: [n][k]
  const int tid  = threadIdx.x;
  const int lane = tid & 31;
  const int wave = tid >> 5;
  const int m0 = blockIdx.y * 64;
  const int n0 = blockIdx.x * 64;
  const int wr = (wave >> 1) * 32;
  const int wc = (wave & 1) * 32;

  v8f acc[2][2];
#pragma unroll
  for (int i = 0; i < 2; ++i)
#pragma unroll
    for (int j = 0; j < 2; ++j) acc[i][j] = vzero8();

  for (int k0 = 0; k0 < EMBED; k0 += 32) {
    __syncthreads();
#pragma unroll
    for (int i = 0; i < 4; ++i) {        // A tile 64x32 bf16, packed copies
      int e = tid + i * 128;
      int r = e >> 3, q = (e & 7) * 4;
      *(u32x2*)(&As[r * LDK32 + q]) =
          *(const u32x2*)(ctx + (size_t)(m0 + r) * EMBED + k0 + q);
    }
    // B tile 32x64 f32 -> transposed bf16; pair along k -> packed b32 stores
#pragma unroll
    for (int i = 0; i < 8; ++i) {
      int e = tid + i * 128;
      int k = (e >> 6) * 2, n = e & 63;
      const float* wp = Wp + (size_t)(k0 + k) * EMBED + n0 + n;
      Pack2 w;
      w.h[0] = (__bf16)wp[0];
      w.h[1] = (__bf16)wp[EMBED];
      *(unsigned int*)(&Bs[n * LDK32 + k]) = w.u;
    }
    __syncthreads();

    v16bf a0 = load_a_frag(As, LDK32, wr + 0, 0, lane);
    v16bf a1 = load_a_frag(As, LDK32, wr + 16, 0, lane);
    v16bf b0 = load_bT_frag(Bs, LDK32, 0, wc + 0, lane);
    v16bf b1 = load_bT_frag(Bs, LDK32, 0, wc + 16, lane);
    acc[0][0] = wmma_bf16(a0, b0, acc[0][0]);
    acc[0][1] = wmma_bf16(a0, b1, acc[0][1]);
    acc[1][0] = wmma_bf16(a1, b0, acc[1][0]);
    acc[1][1] = wmma_bf16(a1, b1, acc[1][1]);
  }

#pragma unroll
  for (int i = 0; i < 2; ++i)
#pragma unroll
    for (int j = 0; j < 2; ++j)
#pragma unroll
      for (int r = 0; r < 8; ++r) {
        int row = m0 + wr + 16 * i + r + ((lane & 16) ? 8 : 0);
        int col = n0 + wc + 16 * j + (lane & 15);
        out[(size_t)row * EMBED + col] = acc[i][j][r] + bp[col];
      }
}

// ---------------------------------------------------------------------------
// Launch: x, W_qkv, b_qkv, W_proj, b_proj -> out (f32 [2,2048,1024])
// Workspace: Q,K,V each [2,16,2048,64] bf16 + ctx [2,2048,1024] bf16 = 32 MB.
// ---------------------------------------------------------------------------
extern "C" void kernel_launch(void* const* d_in, const int* in_sizes, int n_in,
                              void* d_out, int out_size, void* d_ws, size_t ws_size,
                              hipStream_t stream) {
  (void)in_sizes; (void)n_in; (void)out_size; (void)ws_size;
  const float* x    = (const float*)d_in[0];
  const float* Wqkv = (const float*)d_in[1];
  const float* bqkv = (const float*)d_in[2];
  const float* Wp   = (const float*)d_in[3];
  const float* bp   = (const float*)d_in[4];
  float* out = (float*)d_out;

  const size_t qkv_elems = (size_t)BATCH * HEADS * SEQ * HDIM;
  __bf16* qg  = (__bf16*)d_ws;
  __bf16* kg  = qg + qkv_elems;
  __bf16* vg  = kg + qkv_elems;
  __bf16* ctx = vg + qkv_elems;

  qkv_gemm_kernel<<<dim3(QKVN / 64, MTOT / 64), 128, 0, stream>>>(x, Wqkv, bqkv, qg, kg, vg);
  attn_kernel<<<dim3(SEQ / 64, BATCH * HEADS), 128, 0, stream>>>(qg, kg, vg, ctx);
  proj_gemm_kernel<<<dim3(EMBED / 64, MTOT / 64), 128, 0, stream>>>(ctx, Wp, bp, out);
}
